// baseline_attention_31464930411224
// MI455X (gfx1250) — compile-verified, matrix operand order checked by reading, not run on hardware
//
#include <hip/hip_runtime.h>
#include <hip/hip_bf16.h>
#include <math.h>

// Problem constants (from reference)
#define B_    32
#define REP_  1024
#define NV_   504
#define NR_   190
#define NN_   11538
#define MR_   6
#define HID_  32
#define NH_   4
#define MAXN_ 512

typedef _Float16 v16h __attribute__((ext_vector_type(16)));
typedef float    v8f  __attribute__((ext_vector_type(8)));

// ---------------------------------------------------------------------------
// 16x64-tile WMMA GEMM, one wave per workgroup, 4 back-to-back WMMAs per
// K-step reusing one staged A fragment (operand reuse; K=32 per step).
//   C[MxN] = A[MxK] (row-major, lda) @ B + bias
//   BT=0: B row-major  B[k*ldb + n]
//   BT=1: B transposed B[n*ldb + k]   (for  x @ W.T)
//   STN=0: C row-major C[m*ldc + n]
//   STN=1: "node" scatter store: m=batch(0..31), n=role*32+hid ->
//          C[(n>>5)*1024 + m*32 + (n&31)]   (node laid out (L, B, HID))
// fp32 inputs are staged through LDS as fp16 with float2 (global_load_b64)
// vector loads (all ld's even -> 8B aligned). LDS B layout is [n][k] so
// fragment reads are contiguous K-pairs (mergeable 16-bit DS reads).
// Fragments follow ISA 7.12.2; EXEC is all-ones around the WMMAs.
// ---------------------------------------------------------------------------
template <int BT, int STN>
__global__ __launch_bounds__(32)
void wmma_gemm_f16(const float* __restrict__ A, const float* __restrict__ Bm,
                   const float* __restrict__ bias, float* __restrict__ C,
                   int M, int N, int K, int lda, int ldb, int ldc)
{
    __shared__ _Float16 sA[16 * 32];   // [row][k]
    __shared__ _Float16 sB[64 * 32];   // [n_local][k]

    const int lane = threadIdx.x;      // 0..31
    const int n0 = blockIdx.x * 64;
    const int m0 = blockIdx.y * 16;
    const int r  = lane & 15;
    const int h  = lane >> 4;

    v8f acc0 = {}, acc1 = {}, acc2 = {}, acc3 = {};

    for (int kk = 0; kk < K; kk += 32) {
        // ---- stage A tile: 16 rows x 32 k, float2 loads (M multiple of 16)
        #pragma unroll
        for (int i = 0; i < 8; ++i) {
            int e2  = lane + 32 * i;             // 0..255 pairs
            int row = e2 >> 4, k = (e2 & 15) * 2;
            const float2 g = *(const float2*)&A[(size_t)(m0 + row) * lda + kk + k];
            sA[row * 32 + k]     = (_Float16)g.x;
            sA[row * 32 + k + 1] = (_Float16)g.y;
        }
        // ---- stage B tile: 32 k x 64 cols into sB[n][k]
        #pragma unroll
        for (int i = 0; i < 32; ++i) {
            int e2 = lane + 32 * i;              // 0..1023 pairs
            if (BT == 0) {
                int k = e2 >> 5, c2 = e2 & 31;   // pair of adjacent columns
                int col = n0 + c2 * 2; if (col > N - 2) col = N - 2;
                const float2 g = *(const float2*)&Bm[(size_t)(kk + k) * ldb + col];
                sB[(c2 * 2) * 32 + k]     = (_Float16)g.x;
                sB[(c2 * 2 + 1) * 32 + k] = (_Float16)g.y;
            } else {
                int nl = e2 >> 4, k = (e2 & 15) * 2;  // pair of adjacent k
                int col = n0 + nl; if (col > N - 1) col = N - 1;
                const float2 g = *(const float2*)&Bm[(size_t)col * ldb + kk + k];
                sB[nl * 32 + k]     = (_Float16)g.x;
                sB[nl * 32 + k + 1] = (_Float16)g.y;
            }
        }
        // prefetch next K tile into cache (global_prefetch_b8)
        if (kk + 32 < K) {
            __builtin_prefetch(&A[(size_t)(m0 + r) * lda + kk + 32], 0, 0);
            if (BT == 0) __builtin_prefetch(&Bm[(size_t)(kk + 32) * ldb + n0 + lane * 2], 0, 0);
        }
        __syncthreads();

        // ---- A fragment (built once, reused by 4 WMMAs)
        v16h a;
        #pragma unroll
        for (int i = 0; i < 8; ++i) {
            int k = (i < 4) ? (2 * i + 8 * h) : (16 + 2 * (i - 4) + 8 * h);
            a[2 * i]     = sA[r * 32 + k];
            a[2 * i + 1] = sA[r * 32 + k + 1];
        }
        // ---- 4 N-subtiles
        #pragma unroll
        for (int j = 0; j < 4; ++j) {
            v16h b;
            const int nl = j * 16 + r;
            #pragma unroll
            for (int i = 0; i < 8; ++i) {
                int k = (i < 4) ? (2 * i + 8 * h) : (16 + 2 * (i - 4) + 8 * h);
                b[2 * i]     = sB[nl * 32 + k];
                b[2 * i + 1] = sB[nl * 32 + k + 1];
            }
            if (j == 0) acc0 = __builtin_amdgcn_wmma_f32_16x16x32_f16(false, a, false, b, (short)0, acc0, false, false);
            if (j == 1) acc1 = __builtin_amdgcn_wmma_f32_16x16x32_f16(false, a, false, b, (short)0, acc1, false, false);
            if (j == 2) acc2 = __builtin_amdgcn_wmma_f32_16x16x32_f16(false, a, false, b, (short)0, acc2, false, false);
            if (j == 3) acc3 = __builtin_amdgcn_wmma_f32_16x16x32_f16(false, a, false, b, (short)0, acc3, false, false);
        }
        __syncthreads();
    }

    // ---- store: C vgpr jj -> M = m0 + jj + 8h, N = n0 + j*16 + (lane&15)
    #pragma unroll
    for (int j = 0; j < 4; ++j) {
        const v8f acc = (j == 0) ? acc0 : (j == 1) ? acc1 : (j == 2) ? acc2 : acc3;
        int n = n0 + j * 16 + r;
        if (n < N) {
            float bv = bias ? bias[n] : 0.0f;
            #pragma unroll
            for (int jj = 0; jj < 8; ++jj) {
                int m = m0 + jj + 8 * h;
                float val = acc[jj] + bv;
                if (STN == 0)
                    C[(size_t)m * ldc + n] = val;
                else
                    C[(size_t)(n >> 5) * 1024 + (size_t)m * 32 + (n & 31)] = val;
            }
        }
    }
}

// ---------------------------------------------------------------------------
// Attention core: one block per (batch, head); thread t owns query row t.
// qkv: (L, B, 96) with q=[0,32) k=[32,64) v=[64,96); att: (L, B, 32).
// ---------------------------------------------------------------------------
__global__ __launch_bounds__(192)
void mha_attn(const float* __restrict__ qkv, float* __restrict__ att)
{
    __shared__ float qs[NR_ * 8], ks[NR_ * 8], vs[NR_ * 8];
    const int bh = blockIdx.x;          // 0..127
    const int b = bh >> 2, hh = bh & 3;
    const int t = threadIdx.x;

    for (int e = t; e < NR_ * 8; e += blockDim.x) {
        int l = e >> 3, d = e & 7;
        size_t base = ((size_t)l * B_ + b) * 96 + hh * 8 + d;
        qs[e] = qkv[base];
        ks[e] = qkv[base + 32];
        vs[e] = qkv[base + 64];
    }
    __syncthreads();

    if (t < NR_) {
        const float scale = 0.35355339059327373f;  // 1/sqrt(8)
        float q[8];
        #pragma unroll
        for (int d = 0; d < 8; ++d) q[d] = qs[t * 8 + d];

        float mx = -3.4e38f;
        for (int m = 0; m < NR_; ++m) {
            float s = 0.f;
            #pragma unroll
            for (int d = 0; d < 8; ++d) s += q[d] * ks[m * 8 + d];
            s *= scale;
            if (s > mx) mx = s;
        }
        float sum = 0.f, accv[8] = {0, 0, 0, 0, 0, 0, 0, 0};
        for (int m = 0; m < NR_; ++m) {
            float s = 0.f;
            #pragma unroll
            for (int d = 0; d < 8; ++d) s += q[d] * ks[m * 8 + d];
            float e = expf(s * scale - mx);
            sum += e;
            #pragma unroll
            for (int d = 0; d < 8; ++d) accv[d] += e * vs[m * 8 + d];
        }
        float inv = 1.0f / sum;
        #pragma unroll
        for (int d = 0; d < 8; ++d)
            att[((size_t)t * B_ + b) * HID_ + hh * 8 + d] = accv[d] * inv;
    }
}

// ---------------------------------------------------------------------------
// Per-(role, batch) gather + online logsumexp/max/argmax over 512 nouns.
// Writes into (B, 191) arrays with column 0 = zeros (the "null role").
// ---------------------------------------------------------------------------
__global__ void grouped_lse(const float* __restrict__ rn, const int* __restrict__ ridx,
                            float* __restrict__ marg, float* __restrict__ rmax,
                            int* __restrict__ rmaxi)
{
    int id = blockIdx.x * blockDim.x + threadIdx.x;
    if (id >= B_ * (NR_ + 1)) return;
    int b = id / (NR_ + 1), c = id % (NR_ + 1);
    if (c == 0) {
        marg[b * (NR_ + 1)] = 0.f; rmax[b * (NR_ + 1)] = 0.f; rmaxi[b * (NR_ + 1)] = 0;
        return;
    }
    int l = c - 1;
    const int*   idxp = ridx + (size_t)l * MAXN_;
    const float* pot  = rn + ((size_t)l * B_ + b) * NN_;
    float mx = -3.4e38f, s = 0.f; int aj = 0;
    for (int j = 0; j < MAXN_; ++j) {
        float v = pot[idxp[j]];
        if (v > mx) { s = s * expf(mx - v) + 1.0f; aj = j; mx = v; }
        else        { s += expf(v - mx); }
    }
    marg[b * (NR_ + 1) + c]  = mx + logf(s);
    rmax[b * (NR_ + 1) + c]  = mx;
    rmaxi[b * (NR_ + 1) + c] = idxp[aj];
}

// ---------------------------------------------------------------------------
// Per-(batch, verb): gather 6 roles, sum marginals/maxes, emit maxi indices.
// ---------------------------------------------------------------------------
__global__ void verb_kernel(const float* __restrict__ marg, const float* __restrict__ rmax,
                            const int* __restrict__ rmaxi, const int* __restrict__ pad,
                            const float* __restrict__ vpot,
                            float* __restrict__ vmarg, float* __restrict__ vmax_out,
                            int* __restrict__ maxi_out)
{
    int id = blockIdx.x * blockDim.x + threadIdx.x;
    if (id >= B_ * NV_) return;
    int b = id / NV_, v = id % NV_;
    float ms = 0.f, xs = 0.f;
    #pragma unroll
    for (int r = 0; r < MR_; ++r) {
        int p = pad[v * MR_ + r];                     // 0..190
        ms += marg[b * (NR_ + 1) + p];
        xs += rmax[b * (NR_ + 1) + p];
        maxi_out[((size_t)b * NV_ + v) * MR_ + r] = rmaxi[b * (NR_ + 1) + p];
    }
    float vp = vpot[(size_t)b * NV_ + v];
    vmarg[(size_t)b * NV_ + v]    = ms + vp;
    vmax_out[(size_t)b * NV_ + v] = xs + vp;
}

// Per-batch logsumexp over 504 verbs -> norm
__global__ void norm_kernel(const float* __restrict__ vmarg, float* __restrict__ norm)
{
    int b = threadIdx.x;
    if (b >= B_) return;
    float mx = -3.4e38f;
    for (int v = 0; v < NV_; ++v) { float x = vmarg[b * NV_ + v]; if (x > mx) mx = x; }
    float s = 0.f;
    for (int v = 0; v < NV_; ++v) s += expf(vmarg[b * NV_ + v] - mx);
    norm[b] = mx + logf(s);
}

// ---------------------------------------------------------------------------
extern "C" void kernel_launch(void* const* d_in, const int* in_sizes, int n_in,
                              void* d_out, int out_size, void* d_ws, size_t ws_size,
                              hipStream_t stream)
{
    (void)in_sizes; (void)n_in; (void)out_size; (void)ws_size;

    const float* rep    = (const float*)d_in[0];
    const float* W_v    = (const float*)d_in[1];
    const float* b_v    = (const float*)d_in[2];
    const float* W_r    = (const float*)d_in[3];
    const float* b_r    = (const float*)d_in[4];
    const float* ain_w  = (const float*)d_in[5];   // (3, 96, 32)
    const float* ain_b  = (const float*)d_in[6];   // (3, 96)
    const float* aout_w = (const float*)d_in[7];   // (3, 32, 32)
    const float* aout_b = (const float*)d_in[8];   // (3, 32)
    const float* W_n    = (const float*)d_in[9];
    const float* b_n    = (const float*)d_in[10];
    const int*   ridx   = (const int*)d_in[11];
    const int*   pad    = (const int*)d_in[12];

    float* out = (float*)d_out;
    const size_t O1 = (size_t)B_ * REP_;                  // v_potential
    const size_t O2 = O1 + (size_t)B_ * NV_;              // rn_potential
    const size_t O3 = O2 + (size_t)NR_ * B_ * NN_;        // norm
    const size_t O4 = O3 + B_;                            // v_max
    const size_t O5 = O4 + (size_t)B_ * NV_;              // vr_maxi_g (int32)
    float* v_pot  = out + O1;
    float* rn_pot = out + O2;
    float* norm   = out + O3;
    float* vmax   = out + O4;
    int*   maxi   = (int*)(out + O5);

    float* ws = (float*)d_ws;
    float* node_a = ws;  ws += (size_t)NR_ * B_ * HID_;
    float* node_b = ws;  ws += (size_t)NR_ * B_ * HID_;
    float* qkv    = ws;  ws += (size_t)NR_ * B_ * 96;
    float* att    = ws;  ws += (size_t)NR_ * B_ * HID_;
    float* marg   = ws;  ws += (size_t)B_ * (NR_ + 1);
    float* rmaxw  = ws;  ws += (size_t)B_ * (NR_ + 1);
    int*   rmaxi  = (int*)ws; ws += (size_t)B_ * (NR_ + 1);
    float* vmarg  = ws;  ws += (size_t)B_ * NV_;

    // out[0]: pass-through copy of rep
    hipMemcpyAsync(d_out, d_in[0], (size_t)B_ * REP_ * sizeof(float),
                   hipMemcpyDeviceToDevice, stream);

    // role projection: (32 x 1024) @ (1024 x 6080) -> node (L,B,HID) scatter
    wmma_gemm_f16<0, 1><<<dim3(95, 2), 32, 0, stream>>>(
        rep, W_r, b_r, node_a, B_, NR_ * HID_, REP_, REP_, NR_ * HID_, 0);
    // verb potential: (32 x 1024) @ (1024 x 504)
    wmma_gemm_f16<0, 0><<<dim3(8, 2), 32, 0, stream>>>(
        rep, W_v, b_v, v_pot, B_, NV_, REP_, REP_, NV_, NV_);

    // 3 MHA layers
    float* cur = node_a; float* nxt = node_b;
    for (int i = 0; i < 3; ++i) {
        // qkv = x @ in_w.T + in_b : M=6080, N=96, K=32, B transposed
        wmma_gemm_f16<1, 0><<<dim3(2, 380), 32, 0, stream>>>(
            cur, ain_w + (size_t)i * 96 * 32, ain_b + (size_t)i * 96, qkv,
            NR_ * B_, 96, HID_, HID_, HID_, 96);
        mha_attn<<<B_ * NH_, 192, 0, stream>>>(qkv, att);
        // out = att @ out_w.T + out_b : M=6080, N=32, K=32
        wmma_gemm_f16<1, 0><<<dim3(1, 380), 32, 0, stream>>>(
            att, aout_w + (size_t)i * 32 * 32, aout_b + (size_t)i * 32, nxt,
            NR_ * B_, HID_, HID_, HID_, HID_, HID_);
        float* t = cur; cur = nxt; nxt = t;
    }

    // noun head: (6080 x 32) @ (32 x 11538) -> 280 MB output (HBM-bound stage)
    wmma_gemm_f16<0, 0><<<dim3((NN_ + 63) / 64, 380), 32, 0, stream>>>(
        cur, W_n, b_n, rn_pot, NR_ * B_, NN_, HID_, HID_, NN_, NN_);

    grouped_lse<<<(B_ * (NR_ + 1) + 255) / 256, 256, 0, stream>>>(
        rn_pot, ridx, marg, rmaxw, rmaxi);
    verb_kernel<<<(B_ * NV_ + 255) / 256, 256, 0, stream>>>(
        marg, rmaxw, rmaxi, pad, v_pot, vmarg, vmax, maxi);
    norm_kernel<<<1, 32, 0, stream>>>(vmarg, norm);
}
